// MultiheadSelfAttention_23055384444970
// MI455X (gfx1250) — matrix operand order repaired and checked by reading, not compiled, not run on hardware
//
#include <hip/hip_runtime.h>

// ---------------------------------------------------------------------------
// Multi-head causal self-attention for MI455X (gfx1250, wave32, WMMA).
// fp32 -> bf16, WMMA bf16 GEMMs (QKV/out proj) with async-LDS B staging,
// flash-attention with v_wmma_f32_16x16x32_bf16 for QK^T, P*V and row sums.
// ---------------------------------------------------------------------------

typedef __bf16 bf16_t;
typedef __attribute__((ext_vector_type(8)))  __bf16 v8bf;
typedef __attribute__((ext_vector_type(16))) __bf16 v16bf;
typedef __attribute__((ext_vector_type(8)))  float  v8f;
typedef int v4i __attribute__((vector_size(16)));

#define D_MODEL   1024
#define NUM_HEADS 16
#define D_K       64
#define BATCH     2
#define SEQ       2048
#define MTOT      (BATCH * SEQ)

#define NEG_BIG (-3.0e38f)

// ds_swizzle patterns (offset = xor<<10 | or<<5 | and)
#define SWZ_X1   0x041f   // xor 1, and 0x1f
#define SWZ_X2   0x081f
#define SWZ_X4   0x101f
#define SWZ_X8   0x201f
#define SWZ_BC16 0x0010   // broadcast lane g*16 within each 16-lane group
#define SWZ(v, imm) \
  __int_as_float(__builtin_amdgcn_ds_swizzle(__float_as_int(v), (imm)))

// Optional CDNA5 async global->LDS path (guarded; falls back cleanly)
#if defined(__has_builtin)
#if __has_builtin(__builtin_amdgcn_global_load_async_to_lds_b128)
#define USE_ASYNC_B 1
#endif
#endif

#define AS1 __attribute__((address_space(1)))
#define AS3 __attribute__((address_space(3)))

__device__ __forceinline__ void wait_asynccnt0() {
#if defined(__has_builtin)
#if __has_builtin(__builtin_amdgcn_s_wait_asynccnt)
  __builtin_amdgcn_s_wait_asynccnt(0);
  return;
#else
  asm volatile("s_wait_asynccnt 0" ::: "memory");
  return;
#endif
#else
  asm volatile("s_wait_asynccnt 0" ::: "memory");
#endif
}

// ---------------------------------------------------------------------------
// WMMA helpers. VGPR layouts used by these helpers (wave32):
//  A (16x32, 16-bit): lane L = row m=L%16; halves 0..15 = K g*16..+15,
//                     g=L/16 (contiguous k, same order as B).
//  B (32x16, 16-bit): lane L = col n=L%16; halves = K g*16..+15 (this code's order, not the ISA table order).
//  C/D (16x16, f32) : vgpr r, lane L -> (row r+8g, col n).
// ---------------------------------------------------------------------------
__device__ __forceinline__ v8f wmma_bf16(v16bf a, v16bf b, v8f c) {
  return __builtin_amdgcn_wmma_f32_16x16x32_bf16(
      false, a, false, b, (short)0, c, false, false);
}

__device__ __forceinline__ v16bf load_frag_a_g(const bf16_t* base, int ld, int lane) {
  const int m = lane & 15;
  const int g = lane >> 4;
  const bf16_t* p = base + (size_t)m * ld + g * 16;
  v8bf lo = *(const v8bf*)(p);
  v8bf hi = *(const v8bf*)(p + 8);
  return __builtin_shufflevector(lo, hi, 0,1,2,3,4,5,6,7,8,9,10,11,12,13,14,15);
}

__device__ __forceinline__ v16bf load_frag_b_g(const bf16_t* base, int ld, int lane) {
  const int n = lane & 15;
  const int g = lane >> 4;
  return *(const v16bf*)(base + (size_t)n * ld + g * 16);
}

// ---------------------------------------------------------------------------
// fp32 -> bf16 convert
// ---------------------------------------------------------------------------
__global__ __launch_bounds__(256) void cvt_f32_bf16(const float* __restrict__ in,
                                                    bf16_t* __restrict__ out, int n) {
  int i = blockIdx.x * blockDim.x + threadIdx.x;
  if (i < n) out[i] = (bf16_t)in[i];
}

// ---------------------------------------------------------------------------
// Y[m,n] = alpha * sum_k A[m,k] * W[n,k]   (A:[M,K] bf16, W:[N,K] bf16)
// 256 threads = 8 waves; wave -> 32 rows x 64 cols (8 WMMA / k-step).
// The 64x32 B tile (shared by all 8 waves) is staged into LDS with
// GLOBAL_LOAD_ASYNC_TO_LDS_B128 when available (ASYNCcnt + barrier).
// TRANS_STORE=true writes out[n*M + m] (produces V^T).
// ---------------------------------------------------------------------------
template <typename OutT, bool TRANS_STORE>
__global__ __launch_bounds__(256) void gemm_xwT(const bf16_t* __restrict__ A,
                                                const bf16_t* __restrict__ W,
                                                OutT* __restrict__ out,
                                                int M, int N, int K, float alpha) {
  const int lane = threadIdx.x & 31;
  const int wid  = __builtin_amdgcn_readfirstlane(threadIdx.x >> 5);
  const int row0 = blockIdx.x * 256 + wid * 32;
  const int col0 = blockIdx.y * 64;
  const int n    = lane & 15;
  const int g    = lane >> 4;

#if USE_ASYNC_B
  __shared__ bf16_t bsh[64][32];                 // 4KB B tile: [ncol][khalf]
  const int tid  = threadIdx.x;
  const int brow = tid >> 2;                     // 0..63 : B row (N dim)
  const int bcol = (tid & 3) * 8;                // 16B chunk within 64B row
  const bf16_t* bsrc = W + (size_t)(col0 + brow) * K + bcol;
#endif

  const bf16_t* A0 = A + (size_t)row0 * K;
  const bf16_t* A1 = A + (size_t)(row0 + 16) * K;
  v8f acc0[4] = {};
  v8f acc1[4] = {};

  for (int k0 = 0; k0 < K; k0 += 32) {
#if USE_ASYNC_B
    __syncthreads();                             // previous tile fully consumed
    __builtin_amdgcn_global_load_async_to_lds_b128(
        (AS1 v4i*)(bsrc + k0), (AS3 v4i*)(&bsh[brow][bcol]), 0, 0);
    wait_asynccnt0();
    __syncthreads();                             // tile visible to all waves
#endif
    v16bf a0 = load_frag_a_g(A0 + k0, K, lane);
    v16bf a1 = load_frag_a_g(A1 + k0, K, lane);
#pragma unroll
    for (int j = 0; j < 4; ++j) {
#if USE_ASYNC_B
      v16bf b = *(const v16bf*)(&bsh[j * 16 + n][g * 16]);
#else
      v16bf b = load_frag_b_g(W + (size_t)(col0 + j * 16) * K + k0, K, lane);
#endif
      acc0[j] = wmma_bf16(a0, b, acc0[j]);
      acc1[j] = wmma_bf16(a1, b, acc1[j]);
    }
  }

#pragma unroll
  for (int j = 0; j < 4; ++j) {
#pragma unroll
    for (int r = 0; r < 8; ++r) {
      const int col = col0 + j * 16 + n;
      const int ra  = row0 + g * 8 + r;
      const int rb  = ra + 16;
      const float va = acc0[j][r] * alpha;
      const float vb = acc1[j][r] * alpha;
      if (TRANS_STORE) {
        out[(size_t)col * M + ra] = (OutT)va;
        out[(size_t)col * M + rb] = (OutT)vb;
      } else {
        out[(size_t)ra * N + col] = (OutT)va;
        out[(size_t)rb * N + col] = (OutT)vb;
      }
    }
  }
}

// ---------------------------------------------------------------------------
// Flash attention, causal. One wave = 16 queries x d_k=64, key tiles of 64.
// Only the diagonal subtile (kk == q0) applies the causal mask (scalar
// dispatch); row-sum l_i accumulated with a ones-column WMMA.
// ---------------------------------------------------------------------------
__global__ __launch_bounds__(128) void flash_attn(const bf16_t* __restrict__ Q,
                                                  const bf16_t* __restrict__ Km,
                                                  const bf16_t* __restrict__ VT,
                                                  bf16_t* __restrict__ O) {
  __shared__ bf16_t psh[4][16][72];  // per-wave 16x64 P tile; 144B rows

  const int lane = threadIdx.x & 31;
  const int wid  = __builtin_amdgcn_readfirstlane(threadIdx.x >> 5);
  const int b    = blockIdx.y >> 4;
  const int h    = blockIdx.y & 15;
  const int q0   = (blockIdx.x * 4 + wid) * 16;   // scalar
  const int n    = lane & 15;
  const int g    = lane >> 4;

  const size_t qoff = (size_t)(b * SEQ + q0) * D_MODEL + h * D_K;
  const v16bf qa0 = load_frag_a_g(Q + qoff, D_MODEL, lane);
  const v16bf qa1 = load_frag_a_g(Q + qoff + 32, D_MODEL, lane);

  // Ones-column B fragment: P x ones -> row sums in column n==0.
  v16bf onesb;
#pragma unroll
  for (int i = 0; i < 16; ++i) onesb[i] = (n == 0) ? (bf16_t)1.0f : (bf16_t)0.0f;

  v8f o[4] = {};
  v8f ls = {};
  float mi[8];
#pragma unroll
  for (int r = 0; r < 8; ++r) mi[r] = NEG_BIG;

  const bf16_t* Khead = Km + (size_t)b * SEQ * D_MODEL + h * D_K;

  for (int key0 = 0; key0 <= q0; key0 += 64) {
    v8f s[4];
#pragma unroll
    for (int sub = 0; sub < 4; ++sub) {
      const int kk = key0 + sub * 16;             // scalar
      if (kk < q0) {                              // interior tile: no mask
        const bf16_t* Kbase = Khead + (size_t)kk * D_MODEL;
        v8f t = {};
        t = wmma_bf16(qa0, load_frag_b_g(Kbase, D_MODEL, lane), t);
        t = wmma_bf16(qa1, load_frag_b_g(Kbase + 32, D_MODEL, lane), t);
        s[sub] = t;
      } else if (kk == q0) {                      // diagonal tile: mask once
        const bf16_t* Kbase = Khead + (size_t)kk * D_MODEL;
        v8f t = {};
        t = wmma_bf16(qa0, load_frag_b_g(Kbase, D_MODEL, lane), t);
        t = wmma_bf16(qa1, load_frag_b_g(Kbase + 32, D_MODEL, lane), t);
#pragma unroll
        for (int r = 0; r < 8; ++r) {
          t[r] = (n > (g * 8 + r)) ? NEG_BIG : t[r];
        }
        s[sub] = t;
      } else {                                    // fully masked
#pragma unroll
        for (int r = 0; r < 8; ++r) s[sub][r] = NEG_BIG;
      }
    }

    // Online softmax: row max via ds_swizzle xor-reduce, exp, rescale.
#pragma unroll
    for (int r = 0; r < 8; ++r) {
      float v = fmaxf(fmaxf(s[0][r], s[1][r]), fmaxf(s[2][r], s[3][r]));
      v = fmaxf(v, SWZ(v, SWZ_X1));
      v = fmaxf(v, SWZ(v, SWZ_X2));
      v = fmaxf(v, SWZ(v, SWZ_X4));
      v = fmaxf(v, SWZ(v, SWZ_X8));
      const float mnew = fmaxf(mi[r], v);
      const float sc   = __expf(mi[r] - mnew);
      mi[r] = mnew;
      ls[r] *= sc;
#pragma unroll
      for (int j = 0; j < 4; ++j) o[j][r] *= sc;
#pragma unroll
      for (int sub = 0; sub < 4; ++sub) {
        psh[wid][g * 8 + r][sub * 16 + n] = (bf16_t)__expf(s[sub][r] - mnew);
      }
    }

    asm volatile("s_wait_dscnt 0" ::: "memory");  // cross-lane LDS hand-off

    const bf16_t* prow = &psh[wid][lane & 15][0];
    v8bf plo0 = *(const v8bf*)(prow + g * 16);
    v8bf phi0 = *(const v8bf*)(prow + 8 + g * 16);
    v8bf plo1 = *(const v8bf*)(prow + 32 + g * 16);
    v8bf phi1 = *(const v8bf*)(prow + 40 + g * 16);
    v16bf pa0 = __builtin_shufflevector(plo0, phi0, 0,1,2,3,4,5,6,7,8,9,10,11,12,13,14,15);
    v16bf pa1 = __builtin_shufflevector(plo1, phi1, 0,1,2,3,4,5,6,7,8,9,10,11,12,13,14,15);

    // l_i += P * ones (row sums via WMMA)
    ls = wmma_bf16(pa0, onesb, ls);
    ls = wmma_bf16(pa1, onesb, ls);

    // o += P (16x64) * V (64x64) via V^T row loads
#pragma unroll
    for (int j = 0; j < 4; ++j) {
      const bf16_t* Vbase =
          VT + (size_t)(h * D_K + j * 16) * MTOT + (size_t)b * SEQ + key0;
      v16bf vb0 = load_frag_b_g(Vbase, MTOT, lane);
      v16bf vb1 = load_frag_b_g(Vbase + 32, MTOT, lane);
      o[j] = wmma_bf16(pa0, vb0, o[j]);
      o[j] = wmma_bf16(pa1, vb1, o[j]);
    }
  }

  // Epilogue: broadcast l_i from lane n==0 (ds_swizzle BCASTX16), normalize.
#pragma unroll
  for (int r = 0; r < 8; ++r) {
    const float li  = SWZ(ls[r], SWZ_BC16);
    const float inv = 1.0f / li;
#pragma unroll
    for (int j = 0; j < 4; ++j) {
      const size_t idx =
          (size_t)(b * SEQ + q0 + g * 8 + r) * D_MODEL + h * D_K + j * 16 + n;
      O[idx] = (bf16_t)(o[j][r] * inv);
    }
  }
}

// ---------------------------------------------------------------------------
// Launcher
// ---------------------------------------------------------------------------
extern "C" void kernel_launch(void* const* d_in, const int* in_sizes, int n_in,
                              void* d_out, int out_size, void* d_ws, size_t ws_size,
                              hipStream_t stream) {
  (void)in_sizes; (void)n_in; (void)out_size; (void)ws_size;

  const float* x  = (const float*)d_in[0];
  const float* Wq = (const float*)d_in[1];
  const float* Wk = (const float*)d_in[2];
  const float* Wv = (const float*)d_in[3];
  const float* Wo = (const float*)d_in[4];
  float* out = (float*)d_out;

  char* ws = (char*)d_ws;
  size_t off = 0;
  auto alloc = [&](size_t bytes) -> void* {
    void* p = ws + off;
    off += (bytes + 255) & ~(size_t)255;
    return p;
  };

  const size_t nx = (size_t)MTOT * D_MODEL;
  const size_t nw = (size_t)D_MODEL * D_MODEL;

  bf16_t* xb  = (bf16_t*)alloc(nx * 2);
  bf16_t* wqb = (bf16_t*)alloc(nw * 2);
  bf16_t* wkb = (bf16_t*)alloc(nw * 2);
  bf16_t* wvb = (bf16_t*)alloc(nw * 2);
  bf16_t* wob = (bf16_t*)alloc(nw * 2);
  bf16_t* Qb  = (bf16_t*)alloc(nx * 2);
  bf16_t* Kb  = (bf16_t*)alloc(nx * 2);
  bf16_t* VTb = (bf16_t*)alloc(nx * 2);   // [D_MODEL, MTOT]
  bf16_t* Ob  = (bf16_t*)alloc(nx * 2);

  cvt_f32_bf16<<<(int)((nx + 255) / 256), 256, 0, stream>>>(x,  xb,  (int)nx);
  cvt_f32_bf16<<<(int)((nw + 255) / 256), 256, 0, stream>>>(Wq, wqb, (int)nw);
  cvt_f32_bf16<<<(int)((nw + 255) / 256), 256, 0, stream>>>(Wk, wkb, (int)nw);
  cvt_f32_bf16<<<(int)((nw + 255) / 256), 256, 0, stream>>>(Wv, wvb, (int)nw);
  cvt_f32_bf16<<<(int)((nw + 255) / 256), 256, 0, stream>>>(Wo, wob, (int)nw);

  dim3 gg(MTOT / 256, D_MODEL / 64);
  gemm_xwT<bf16_t, false><<<gg, 256, 0, stream>>>(xb, wqb, Qb,  MTOT, D_MODEL, D_MODEL, 0.125f);
  gemm_xwT<bf16_t, false><<<gg, 256, 0, stream>>>(xb, wkb, Kb,  MTOT, D_MODEL, D_MODEL, 1.0f);
  gemm_xwT<bf16_t, true ><<<gg, 256, 0, stream>>>(xb, wvb, VTb, MTOT, D_MODEL, D_MODEL, 1.0f);

  flash_attn<<<dim3(SEQ / 64, BATCH * NUM_HEADS), 128, 0, stream>>>(Qb, Kb, VTb, Ob);

  gemm_xwT<float, false><<<gg, 256, 0, stream>>>(Ob, wob, out, MTOT, D_MODEL, D_MODEL, 1.0f);
}
